// CrystalGraphConvNet_1709396984469
// MI455X (gfx1250) — matrix operand order repaired and checked by reading, not compiled, not run on hardware
//
#include <hip/hip_runtime.h>
#include <math.h>

#define Natoms 20000
#define Mnbr   12
#define NMROWS 240000
#define Ffeat  128
#define C2     256       // 2F
#define ORIGF  92
#define NBRF   64
#define SLOPE  0.01f
#define BN_EPS 1e-5f
#define STEPSZ 0.3f

typedef __attribute__((ext_vector_type(16))) __bf16 v16bf;
typedef __attribute__((ext_vector_type(8)))  float  v8f;

__device__ __forceinline__ unsigned short f2bf(float f){
  unsigned u = __float_as_uint(f);
  u += 0x7FFFu + ((u >> 16) & 1u);          // round-to-nearest-even
  return (unsigned short)(u >> 16);
}
__device__ __forceinline__ float leakyf(float x){ return x >= 0.f ? x : SLOPE * x; }
__device__ __forceinline__ float sigmoidf(float x){ return 1.f / (1.f + expf(-x)); }
__device__ __forceinline__ float softplusf(float x){ return x > 20.f ? x : log1pf(expf(x)); }

// ---------------- prep kernels ----------------

// x = atom_fea @ emb_W + emb_b  (and bf16 copy for WMMA)
__global__ void k_embed(const float* __restrict__ af, const float* __restrict__ Wm,
                        const float* __restrict__ bb, float* __restrict__ x,
                        unsigned short* __restrict__ xbf){
  int t = blockIdx.x * 256 + threadIdx.x;   // exactly N*F threads
  int n = t >> 7, f = t & 127;
  float acc = bb[f];
  const float* ar = af + n * ORIGF;
  #pragma unroll 4
  for (int k = 0; k < ORIGF; ++k) acc += ar[k] * Wm[k * Ffeat + f];
  x[t] = acc;
  xbf[t] = f2bf(acc);
}

__global__ void k_nbr2bf(const float* __restrict__ nf, unsigned short* __restrict__ nb){
  int t = blockIdx.x * 256 + threadIdx.x;   // exactly N*M*64
  nb[t] = f2bf(nf[t]);
}

// conv_W (3 x 320 x 256 f32) -> WMMA B-fragment-ready bf16 layout
// element ((ct*10+kc)*32 + lane)*16 + h  =  W[k = kc*32 + (lane>>4)*8 + (h&7) + (h>>3)*16][col = ct*16 + (lane&15)]
__global__ void k_wfrag(const float* __restrict__ cw, unsigned short* __restrict__ wf){
  int t = blockIdx.x * 256 + threadIdx.x;   // exactly 3*81920
  int l = t / 81920, r = t % 81920;
  int h = r & 15, e = r >> 4;
  int lane = e & 31, blk = e >> 5;
  int kc = blk % 10, ct = blk / 10;
  int col = ct * 16 + (lane & 15);
  int k   = kc * 32 + (lane >> 4) * 8 + (h & 7) + (h >> 3) * 16;
  wf[t] = f2bf(cw[l * 81920 + k * C2 + col]);
}

__global__ void k_zero(float* p, int n){
  int t = blockIdx.x * 256 + threadIdx.x;
  if (t < n) p[t] = 0.f;
}

// ---------------- WMMA conv GEMM ----------------
// Workgroup = 128 threads (4 waves), 64 rows x 256 cols.  z = A(64x320 bf16) * W(320x256 bf16) + bias,
// fused per-channel sum / sumsq for train-mode BN1 (LDS reduce, then global atomics).
__global__ __launch_bounds__(128) void k_conv(
    const unsigned short* __restrict__ xbf,
    const unsigned short* __restrict__ nbrbf,
    const int*            __restrict__ idx,
    const unsigned short* __restrict__ wfrag,
    const float*          __restrict__ bias,
    float*                __restrict__ z,
    float* __restrict__ bsum, float* __restrict__ bsq)
{
  __shared__ __align__(16) unsigned short ldsA[64 * 328];   // 64 rows x 320 halves (pad 8)
  __shared__ float ldsS[C2];
  __shared__ float ldsQ[C2];
  const int tid  = threadIdx.x;
  const int wave = tid >> 5, lane = tid & 31;
  const int mrow = lane & 15, hi = lane >> 4;
  const long rowBase = (long)blockIdx.x * 64;               // 3750 blocks -> 240000 rows exact

  for (int i = tid; i < C2; i += 128){ ldsS[i] = 0.f; ldsQ[i] = 0.f; }

  // stage gathered A tile: [self x | neighbor x | edge fea] per row, 16B chunks
  for (int c = tid; c < 64 * 40; c += 128){
    int r = c / 40;
    int h = (c % 40) * 8;                                   // half index within row
    long row = rowBase + r;
    int n = (int)(row / Mnbr), m = (int)(row % Mnbr);
    const unsigned short* src;
    if (h < Ffeat)            src = xbf + (long)n * Ffeat + h;
    else if (h < 2 * Ffeat) { int j = idx[n * Mnbr + m];
                              src = xbf + (long)j * Ffeat + (h - Ffeat); }
    else                      src = nbrbf + ((long)n * Mnbr + m) * NBRF + (h - 2 * Ffeat);
    *(uint4*)&ldsA[r * 328 + h] = *(const uint4*)src;
  }
  __syncthreads();

  // A fragments for this wave's 16 rows, all 10 K-chunks kept in registers (80 VGPRs)
  v16bf a[10];
  #pragma unroll
  for (int kc = 0; kc < 10; ++kc){
    const unsigned short* p = &ldsA[(wave * 16 + mrow) * 328 + kc * 32 + hi * 8];
    union { uint4 u[2]; v16bf v; } U;
    U.u[0] = *(const uint4*)p;          // K = base + 0..7   (lane-half selected)
    U.u[1] = *(const uint4*)(p + 16);   // K = base + 16..23
    a[kc] = U.v;
  }

  const v16bf* wb = (const v16bf*)wfrag;
  for (int ct = 0; ct < 16; ++ct){
    int col = ct * 16 + mrow;
    float bv = bias[col];
    v8f acc = {bv, bv, bv, bv, bv, bv, bv, bv};
    #pragma unroll
    for (int kc = 0; kc < 10; ++kc){
      v16bf b = wb[(ct * 10 + kc) * 32 + lane];
      acc = __builtin_amdgcn_wmma_f32_16x16x32_bf16(false, a[kc], false, b,
                                                    (short)0, acc, false, false);
    }
    float* zp = z + (rowBase + wave * 16 + hi * 8) * (long)C2 + col;
    float s = 0.f, q = 0.f;
    #pragma unroll
    for (int i = 0; i < 8; ++i){
      float v = acc[i];
      zp[(long)i * C2] = v;
      s += v; q += v * v;
    }
    atomicAdd(&ldsS[col], s);
    atomicAdd(&ldsQ[col], q);
  }
  __syncthreads();
  for (int i = tid; i < C2; i += 128){
    atomicAdd(&bsum[i], ldsS[i]);
    atomicAdd(&bsq[i],  ldsQ[i]);
  }
}

// sums -> (scale, shift) for BN
__global__ void k_bnfin(const float* s, const float* q, const float* g, const float* b,
                        float count, float* scale, float* shift, int C){
  int t = threadIdx.x;
  if (t < C){
    float m  = s[t] / count;
    float v  = q[t] / count - m * m;
    float sc = g[t] * rsqrtf(v + BN_EPS);
    scale[t] = sc;
    shift[t] = b[t] - m * sc;
  }
}

// BN1 + sigmoid(filt)*leaky(core), sum over M; fused BN2 channel sums
__global__ void k_reduce_m(const float* __restrict__ z, const float* __restrict__ sc1,
                           const float* __restrict__ sh1, float* __restrict__ summed,
                           float* __restrict__ s2, float* __restrict__ q2){
  __shared__ float lS[Ffeat], lQ[Ffeat];
  int tid = threadIdx.x;
  if (tid < Ffeat){ lS[tid] = 0.f; lQ[tid] = 0.f; }
  __syncthreads();
  int t = blockIdx.x * 256 + tid;            // exactly N*F threads
  int n = t >> 7, f = t & 127;
  float scf = sc1[f], shf = sh1[f], scc = sc1[Ffeat + f], shc = sh1[Ffeat + f];
  float acc = 0.f;
  const float* zr = z + (long)n * Mnbr * C2;
  #pragma unroll
  for (int m = 0; m < Mnbr; ++m){
    float zf = zr[m * C2 + f] * scf + shf;
    float zc = zr[m * C2 + Ffeat + f] * scc + shc;
    acc += sigmoidf(zf) * leakyf(zc);
  }
  summed[t] = acc;
  atomicAdd(&lS[f], acc);
  atomicAdd(&lQ[f], acc * acc);
  __syncthreads();
  if (tid < Ffeat){ atomicAdd(&s2[tid], lS[tid]); atomicAdd(&q2[tid], lQ[tid]); }
}

// x = leaky(x + BN2(summed)); refresh bf16 copy
__global__ void k_bn2res(float* __restrict__ x, const float* __restrict__ summed,
                         const float* __restrict__ sc, const float* __restrict__ sh,
                         unsigned short* __restrict__ xbf){
  int t = blockIdx.x * 256 + threadIdx.x;    // exactly N*F
  int f = t & 127;
  float v = leakyf(x[t] + summed[t] * sc[f] + sh[f]);
  x[t] = v;
  xbf[t] = f2bf(v);
}

// ---------------- bond heads ----------------

// per-atom partial dots x . fd_W/fk_W halves
__global__ void k_partials(const float* __restrict__ x, const float* __restrict__ fd,
                           const float* __restrict__ fk, float* pds, float* pdn,
                           float* pks, float* pkn){
  int t = blockIdx.x * 256 + threadIdx.x;
  if (t >= Natoms) return;
  const float* xr = x + t * Ffeat;
  float a = 0, b = 0, c = 0, d = 0;
  #pragma unroll 4
  for (int i = 0; i < Ffeat; ++i){
    float xv = xr[i];
    a += xv * fd[i];          b += xv * fd[Ffeat + i];
    c += xv * fk[i];          d += xv * fk[Ffeat + i];
  }
  pds[t] = a; pdn[t] = b; pks[t] = c; pkn[t] = d;
}

__global__ void k_bond(const float* __restrict__ noff, const float* __restrict__ cells,
                       const float* __restrict__ apos, const float* __restrict__ npos,
                       const float* __restrict__ nfea, const int* __restrict__ idx,
                       const float* __restrict__ fdW, const float* __restrict__ fdb,
                       const float* __restrict__ fkW, const float* __restrict__ fkb,
                       const float* __restrict__ pds, const float* __restrict__ pdn,
                       const float* __restrict__ pks, const float* __restrict__ pkn,
                       float* __restrict__ offsh, float* __restrict__ bondd,
                       float* __restrict__ bondk){
  int t = blockIdx.x * 256 + threadIdx.x;
  if (t >= NMROWS) return;
  int n = t / Mnbr;
  const float* o  = noff + t * 3;
  const float* c9 = cells + n * 9;
  float o0 = o[0]*c9[0] + o[1]*c9[3] + o[2]*c9[6];   // off_j = sum_i o_i * cells[i][j]
  float o1 = o[0]*c9[1] + o[1]*c9[4] + o[2]*c9[7];
  float o2 = o[0]*c9[2] + o[1]*c9[5] + o[2]*c9[8];
  offsh[t*3+0] = o0; offsh[t*3+1] = o1; offsh[t*3+2] = o2;
  float dx = npos[t*3+0] - apos[n*3+0] + o0;
  float dy = npos[t*3+1] - apos[n*3+1] + o1;
  float dz = npos[t*3+2] - apos[n*3+2] + o2;
  float d0 = sqrtf(dx*dx + dy*dy + dz*dz);
  int j = idx[t];
  float fd = pds[n] + pdn[j] + fdb[0] + d0;
  float fk = pks[n] + pkn[j] + fkb[0] - 4.f;
  const float* nr = nfea + (long)t * NBRF;
  #pragma unroll 4
  for (int i = 0; i < NBRF; ++i){
    float nv = nr[i];
    fd += nv * fdW[2*Ffeat + i];
    fk += nv * fkW[2*Ffeat + i];
  }
  bondd[t] = softplusf(fd);
  bondk[t] = softplusf(fk);
}

// ---------------- device-driven relaxation loop ----------------
// scal[0] = last_max (f32); scal(u32)[1] = cur_max bits; scal(u32)[2] = cont flag

__global__ void k_initrelax(const float* __restrict__ apos, float* __restrict__ pos,
                            float* __restrict__ gout, float* __restrict__ scal){
  int t = blockIdx.x * 256 + threadIdx.x;
  if (t < Natoms * 3){ pos[t] = apos[t]; gout[t] = 0.f; }
  if (t == 0) scal[0] = 1.0f;
}

__global__ void k_iterbegin(float* scal, int iter){
  unsigned* u = (unsigned*)scal;
  u[2] = (scal[0] > 1e-3f || iter < 5) ? 1u : 0u;
  u[1] = 0u;
}

__global__ void k_accum(const float* __restrict__ pos, const float* __restrict__ npos0,
                        const int* __restrict__ idx, const float* __restrict__ offsh,
                        const float* __restrict__ bondd, const float* __restrict__ bondk,
                        float* __restrict__ grad, int first){
  int t = blockIdx.x * 256 + threadIdx.x;
  if (t >= NMROWS) return;
  int n = t / Mnbr;
  int j = idx[t];
  float nx, ny, nz;
  if (first){ nx = npos0[t*3]; ny = npos0[t*3+1]; nz = npos0[t*3+2]; }
  else      { nx = pos[j*3];   ny = pos[j*3+1];   nz = pos[j*3+2]; }
  float vx = nx - pos[n*3]   + offsh[t*3];
  float vy = ny - pos[n*3+1] + offsh[t*3+1];
  float vz = nz - pos[n*3+2] + offsh[t*3+2];
  float d = sqrtf(vx*vx + vy*vy + vz*vz);
  float c = (d > 0.f) ? 2.f * bondk[t] * (bondd[t] - d) / d : 0.f;
  float gx = c*vx, gy = c*vy, gz = c*vz;
  atomicAdd(&grad[n*3],   gx);
  atomicAdd(&grad[n*3+1], gy);
  atomicAdd(&grad[n*3+2], gz);
  if (!first){                               // scatter to neighbor for i>0
    atomicAdd(&grad[j*3],   -gx);
    atomicAdd(&grad[j*3+1], -gy);
    atomicAdd(&grad[j*3+2], -gz);
  }
}

__global__ void k_apply(float* __restrict__ pos, const float* __restrict__ grad,
                        float* __restrict__ gout, const int* __restrict__ fixed,
                        float* __restrict__ scal){
  int t = blockIdx.x * 256 + threadIdx.x;
  if (t >= Natoms) return;
  unsigned cont = ((const unsigned*)scal)[2];
  float fr = 1.f - (float)fixed[t];
  float gx = grad[t*3] * fr, gy = grad[t*3+1] * fr, gz = grad[t*3+2] * fr;
  if (cont){
    float sx = -STEPSZ*gx, sy = -STEPSZ*gy, sz = -STEPSZ*gz;
    pos[t*3]   += sx;
    pos[t*3+1] += sy;
    pos[t*3+2] += sz;
    gout[t*3] = gx; gout[t*3+1] = gy; gout[t*3+2] = gz;
    float ms = fmaxf(fabsf(sx), fmaxf(fabsf(sy), fabsf(sz)));
    atomicMax((unsigned*)scal + 1, __float_as_uint(ms));   // nonneg f32: uint order == f32 order
  }
}

__global__ void k_iterend(float* scal){
  unsigned* u = (unsigned*)scal;
  if (u[2]) scal[0] = __uint_as_float(u[1]);
}

__global__ void k_writeout(const float* __restrict__ pos, const float* __restrict__ gout,
                           float* __restrict__ out){
  int t = blockIdx.x * 256 + threadIdx.x;
  if (t >= Natoms * 3) return;
  out[t]                 = pos[t];
  out[Natoms*3 + t]      = gout[t];
  out[2*Natoms*3 + t]    = pos[t];
}

// ---------------- host launcher ----------------

extern "C" void kernel_launch(void* const* d_in, const int* in_sizes, int n_in,
                              void* d_out, int out_size, void* d_ws, size_t ws_size,
                              hipStream_t stream){
  const float* atom_fea = (const float*)d_in[0];
  const float* nbr_fea  = (const float*)d_in[1];
  const float* nbr_off  = (const float*)d_in[2];
  const float* atom_pos = (const float*)d_in[3];
  const float* nbr_pos  = (const float*)d_in[4];
  const float* cells    = (const float*)d_in[5];
  const float* emb_W    = (const float*)d_in[7];
  const float* emb_b    = (const float*)d_in[8];
  const float* conv_W   = (const float*)d_in[9];
  const float* conv_b   = (const float*)d_in[10];
  const float* bn1_g    = (const float*)d_in[11];
  const float* bn1_b    = (const float*)d_in[12];
  const float* bn2_g    = (const float*)d_in[13];
  const float* bn2_b    = (const float*)d_in[14];
  const float* fd_W     = (const float*)d_in[15];
  const float* fd_b     = (const float*)d_in[16];
  const float* fk_W     = (const float*)d_in[17];
  const float* fk_b     = (const float*)d_in[18];
  const int*   nbr_idx  = (const int*)d_in[19];
  const int*   fixed_m  = (const int*)d_in[22];
  float* out = (float*)d_out;

  char* W = (char*)d_ws;
  float*          x      = (float*)(W + 0UL);            // N*F f32
  float*          summed = (float*)(W + 10240000UL);     // N*F f32
  unsigned short* xbf    = (unsigned short*)(W + 20480000UL);   // N*F bf16
  unsigned short* nbrbf  = (unsigned short*)(W + 25600000UL);   // N*M*64 bf16
  unsigned short* wfrag  = (unsigned short*)(W + 56320000UL);   // 3*81920 bf16 (32B aligned)
  float*          z      = (float*)(W + 56811520UL);     // NM*256 f32
  float*          bn     = (float*)(W + 302571520UL);    // 1536 f32 BN scratch
  float*          pos    = (float*)(W + 302577664UL);    // N*3
  float*          grad   = (float*)(W + 302817664UL);    // N*3
  float*          gout   = (float*)(W + 303057664UL);    // N*3
  float*          offsh  = (float*)(W + 303297664UL);    // NM*3
  float*          bondd  = (float*)(W + 306177664UL);    // NM
  float*          bondk  = (float*)(W + 307137664UL);    // NM
  float*          pds    = (float*)(W + 308097664UL);    // N
  float*          pdn    = (float*)(W + 308177664UL);    // N
  float*          pks    = (float*)(W + 308257664UL);    // N
  float*          pkn    = (float*)(W + 308337664UL);    // N
  float*          scal   = (float*)(W + 308417664UL);    // scalars

  k_embed <<<10000, 256, 0, stream>>>(atom_fea, emb_W, emb_b, x, xbf);
  k_nbr2bf<<<60000, 256, 0, stream>>>(nbr_fea, nbrbf);
  k_wfrag <<<960,   256, 0, stream>>>(conv_W, wfrag);

  for (int l = 0; l < 3; ++l){
    k_zero<<<2, 256, 0, stream>>>(bn, 512);                              // sum1,sq1
    k_conv<<<3750, 128, 0, stream>>>(xbf, nbrbf, nbr_idx, wfrag + l*81920,
                                     conv_b + l*C2, z, bn, bn + 256);
    k_bnfin<<<1, 256, 0, stream>>>(bn, bn + 256, bn1_g + l*C2, bn1_b + l*C2,
                                   (float)NMROWS, bn + 512, bn + 768, C2);
    k_zero<<<1, 256, 0, stream>>>(bn + 1024, 256);                       // sum2,sq2
    k_reduce_m<<<10000, 256, 0, stream>>>(z, bn + 512, bn + 768, summed,
                                          bn + 1024, bn + 1152);
    k_bnfin<<<1, 256, 0, stream>>>(bn + 1024, bn + 1152, bn2_g + l*Ffeat,
                                   bn2_b + l*Ffeat, (float)Natoms,
                                   bn + 1280, bn + 1408, Ffeat);
    k_bn2res<<<10000, 256, 0, stream>>>(x, summed, bn + 1280, bn + 1408, xbf);
  }

  k_partials<<<79, 256, 0, stream>>>(x, fd_W, fk_W, pds, pdn, pks, pkn);
  k_bond<<<938, 256, 0, stream>>>(nbr_off, cells, atom_pos, nbr_pos, nbr_fea, nbr_idx,
                                  fd_W, fd_b, fk_W, fk_b, pds, pdn, pks, pkn,
                                  offsh, bondd, bondk);
  k_initrelax<<<235, 256, 0, stream>>>(atom_pos, pos, gout, scal);

  for (int it = 0; it < 20; ++it){
    k_iterbegin<<<1, 1, 0, stream>>>(scal, it);
    k_zero<<<235, 256, 0, stream>>>(grad, Natoms * 3);
    k_accum<<<938, 256, 0, stream>>>(pos, nbr_pos, nbr_idx, offsh, bondd, bondk,
                                     grad, it == 0 ? 1 : 0);
    k_apply<<<79, 256, 0, stream>>>(pos, grad, gout, fixed_m, scal);
    k_iterend<<<1, 1, 0, stream>>>(scal);
  }

  k_writeout<<<235, 256, 0, stream>>>(pos, gout, out);

  (void)in_sizes; (void)n_in; (void)out_size; (void)ws_size;
}
